// HilbertAttentionTritonFixedOptimized_81630148428359
// MI455X (gfx1250) — matrix operand order repaired and checked by reading, not compiled, not run on hardware
//
#include <hip/hip_runtime.h>
#include <hip/hip_bf16.h>
#include <stdint.h>

// ---------------- problem constants ----------------
#define B_SZ    4
#define M_SZ    4096
#define HID_SZ  1024
#define NH      16
#define HD_SZ   64
#define SEG_SZ  128
#define NSEG    (M_SZ / SEG_SZ)     // 32
#define QKV_N   (3 * HID_SZ)        // 3072
#define ROWS    (B_SZ * M_SZ)       // 16384
#define SCALE_F 0.125f              // HD^-0.5

typedef __attribute__((ext_vector_type(16))) __bf16 v16bf;
typedef __attribute__((ext_vector_type(8)))  float  v8f;
typedef __attribute__((ext_vector_type(4)))  unsigned int v4u;
typedef __attribute__((ext_vector_type(8)))  int v8i;
typedef __attribute__((ext_vector_type(4)))  int v4i;

#if defined(__has_builtin)
#if __has_builtin(__builtin_amdgcn_tensor_load_to_lds)
#define HAVE_TDM 1
#endif
#endif
#ifndef HAVE_TDM
#define HAVE_TDM 0
#endif

union FragU { uint4 u[2]; v16bf v; };

// Two contiguous 16-byte chunks -> one 16x(b16) WMMA operand fragment.
static __device__ __forceinline__ v16bf load_frag(const __bf16* p0, const __bf16* p1) {
    FragU f;
    f.u[0] = *reinterpret_cast<const uint4*>(p0);
    f.u[1] = *reinterpret_cast<const uint4*>(p1);
    return f.v;
}

static __device__ __forceinline__ v8f wmma_bf16(v16bf a, v16bf b, v8f c) {
    return __builtin_amdgcn_wmma_f32_16x16x32_bf16(false, a, false, b, (short)0, c,
                                                   false, false);
}

// ---------------- fp32 -> bf16 convert ----------------
__global__ void cvt_f32_to_bf16(const float* __restrict__ in,
                                __bf16* __restrict__ out, int n) {
    for (int i = blockIdx.x * blockDim.x + threadIdx.x; i < n;
         i += gridDim.x * blockDim.x)
        out[i] = (__bf16)in[i];
}

// ---------------- generic NT GEMM (bf16 in, f32 acc) ----------------
// C[m][n] = sum_k A[m][k] * Bw[n][k].
// Block: 8 waves, tile 128 rows x 64 cols; the 64-col B panel is shared by all
// waves and staged (double-buffered) in LDS by the Tensor Data Mover.
// perm != nullptr: A row gather per 4096-row batch (Hilbert permutation).
// Cf != nullptr -> store fp32, else store bf16 to Cb.
__global__ __launch_bounds__(256) void gemm_nt_bf16(
    const __bf16* __restrict__ A, const __bf16* __restrict__ Bw,
    float* __restrict__ Cf, __bf16* __restrict__ Cb,
    const int* __restrict__ perm, int Mrows, int N, int K) {

    __shared__ alignas(16) __bf16 sB[2][64 * 64];   // 2 x 8 KB B panels

    const int lane = threadIdx.x & 31;
    const int wv   = threadIdx.x >> 5;    // wave 0..7 -> 16-row strip
    const int half = lane >> 4;
    const int l16  = lane & 15;

    const int blocksM = Mrows >> 7;       // 128 rows per block
    const int bm = blockIdx.x % blocksM;
    const int bn = blockIdx.x / blocksM;
    const int n_base = bn * 64;

    // A-fragment source row (row-major, optional Hilbert gather per batch)
    int row = (bm << 7) + (wv << 4) + l16;
    int src_row = row;
    if (perm) {
        int b  = row >> 12;               // 4096 rows per batch
        int ml = row & 4095;
        src_row = (b << 12) + perm[ml];
    }
    const __bf16* arow = A + (size_t)src_row * K;

    const int nstage = K >> 6;            // 64-wide k stages

#if HAVE_TDM
    // Issue one 2-D TDM descriptor: 64 rows x 64 bf16 tile of Bw -> sB[buf].
    // D# group0: count=1 | lds_addr | global_addr[56:0] | type=2
    // D# group1: data_size=2B, tensor_dim0=K, tensor_dim1=64,
    //            tile_dim0=64, tile_dim1=64, tensor_dim0_stride=K
    auto tdm_issue = [&](int stage, int buf) {
        const __bf16* src = Bw + (size_t)n_base * K + (size_t)stage * 64;
        const uint64_t ga = (uint64_t)(uintptr_t)src;
        const uint32_t lds = (uint32_t)(uintptr_t)(&sB[buf][0]); // generic low 32 = LDS offset
        v4u g0 = { 1u,                                  // count=1 (valid user D#)
                   lds,                                 // lds_addr
                   (uint32_t)ga,                        // global_addr[31:0]
                   (uint32_t)((ga >> 32) & 0x1FFFFFFu)  // global_addr[56:32]
                       | (2u << 30) };                  // type=2 ("image")
        v8i g1 = { (int)(1u << 16),                        // data_size=1 (2 bytes)
                   (int)(((uint32_t)K & 0xFFFFu) << 16),   // tensor_dim0[15:0]
                   (int)((((uint32_t)K >> 16) & 0xFFFFu)   // tensor_dim0[31:16]
                       | (64u << 16)),                     // tensor_dim1[15:0]=64
                   (int)(64u << 16),                       // tile_dim0=64
                   64,                                     // tile_dim1=64, tile_dim2=0
                   (int)K,                                 // tensor_dim0_stride[31:0]
                   0, 0 };
        v4i gz4 = { 0, 0, 0, 0 };                          // 2-D tile: groups 2/3 unused
        v8i gz8 = { 0, 0, 0, 0, 0, 0, 0, 0 };              // extra group (unused)
        __builtin_amdgcn_tensor_load_to_lds(g0, g1, gz4, gz4, gz8, 0);
    };
    if (threadIdx.x < 32) tdm_issue(0, 0);
#endif

    v8f acc[4];
    #pragma unroll
    for (int t = 0; t < 4; ++t) acc[t] = (v8f){};

    for (int s = 0; s < nstage; ++s) {
        const int buf = s & 1;
        const int k = s << 6;
#if HAVE_TDM
        if (threadIdx.x < 32) {
            if (s + 1 < nstage) {
                tdm_issue(s + 1, buf ^ 1);              // prefetch next panel
                __builtin_amdgcn_s_wait_tensorcnt(1);   // current panel landed
            } else {
                __builtin_amdgcn_s_wait_tensorcnt(0);
            }
        }
        __syncthreads();
#else
        // cooperative fallback staging: 512 x b128
        for (int lin = threadIdx.x; lin < 512; lin += 256) {
            const int r = lin >> 3, c = lin & 7;
            *reinterpret_cast<uint4*>(&sB[buf][r * 64 + c * 8]) =
                *reinterpret_cast<const uint4*>(&Bw[(size_t)(n_base + r) * K + k + c * 8]);
        }
        __syncthreads();
#endif
        __builtin_prefetch(arow + k + 128, 0, 1);       // global_prefetch_b8
        #pragma unroll
        for (int ks = 0; ks < 2; ++ks) {
            // A 16x32 frag: lanes 0-15 K{0..15}, lanes 16-31 K{16..31}
            v16bf a = load_frag(arow + k + ks * 32 + half * 16,
                                arow + k + ks * 32 + half * 16 + 8);
            #pragma unroll
            for (int t = 0; t < 4; ++t) {
                // B 32x16 frag from LDS panel: lane=col, 16 contiguous K @ half*16
                const __bf16* bp = &sB[buf][(t * 16 + l16) * 64 + ks * 32 + half * 16];
                v16bf b = load_frag(bp, bp + 8);
                acc[t] = wmma_bf16(a, b, acc[t]);
            }
        }
        __syncthreads();   // panel free for the stage-(s+2) TDM write
    }

    // C layout: lane -> col n = l16; VGPR v -> row m0 + v (+8 for upper half)
    const int m0 = (bm << 7) + (wv << 4) + 8 * half;
    #pragma unroll
    for (int t = 0; t < 4; ++t) {
        const int n = n_base + t * 16 + l16;
        #pragma unroll
        for (int v = 0; v < 8; ++v) {
            const size_t r = (size_t)(m0 + v);
            if (Cf) Cf[r * N + n] = acc[t][v];
            else    Cb[r * N + n] = (__bf16)acc[t][v];
        }
    }
}

// ---------------- Hilbert segment attention ----------------
// One block per (b, h, seg). 8 waves; wave w owns score rows [16w, 16w+16).
__global__ __launch_bounds__(256) void hilbert_attn(
    const __bf16* __restrict__ qkv,   // (ROWS, 3072) bf16, Hilbert-permuted rows
    __bf16* __restrict__ y) {         // (ROWS, 1024) bf16 attention output

    __shared__ alignas(16) __bf16 sVT[HD_SZ][SEG_SZ];   // V^T : 16 KB
    __shared__ alignas(16) __bf16 sP[SEG_SZ][SEG_SZ];   // softmax(P) : 32 KB

    const int blk = blockIdx.x;
    const int seg = blk % NSEG;
    const int h   = (blk / NSEG) % NH;
    const int b   = blk / (NSEG * NH);
    const size_t row0 = (size_t)b * M_SZ + (size_t)seg * SEG_SZ;
    const int cq = h * HD_SZ;
    const int ck = HID_SZ + h * HD_SZ;
    const int cv = 2 * HID_SZ + h * HD_SZ;

    // Stage V transposed into LDS: sVT[d][t] = V[t][d]
    for (int idx = threadIdx.x; idx < HD_SZ * SEG_SZ; idx += blockDim.x) {
        const int d = idx >> 7;       // / 128
        const int t = idx & 127;
        sVT[d][t] = qkv[(row0 + t) * QKV_N + cv + d];
    }

    const int lane = threadIdx.x & 31;
    const int half = lane >> 4;
    const int l16  = lane & 15;
    const int s0   = (threadIdx.x >> 5) * 16;   // wave's row tile

    // ---- scores = Q K^T ----
    v8f sc[8];
    #pragma unroll
    for (int t = 0; t < 8; ++t) sc[t] = (v8f){};

    const __bf16* qrow = qkv + (row0 + s0 + l16) * QKV_N + cq;
    #pragma unroll
    for (int k = 0; k < HD_SZ; k += 32) {
        v16bf a = load_frag(qrow + k + half * 16, qrow + k + half * 16 + 8);
        #pragma unroll
        for (int tn = 0; tn < 8; ++tn) {
            const __bf16* krow =
                qkv + (row0 + tn * 16 + l16) * QKV_N + ck + k + half * 16;
            v16bf bb = load_frag(krow, krow + 8);
            sc[tn] = wmma_bf16(a, bb, sc[tn]);
        }
    }

    // ---- scale + dilation mask (DIL=2): drop where s odd AND t odd ----
    #pragma unroll
    for (int tn = 0; tn < 8; ++tn) {
        const int t = tn * 16 + l16;
        #pragma unroll
        for (int v = 0; v < 8; ++v) {
            const int s = s0 + v + 8 * half;
            float val = sc[tn][v] * SCALE_F;
            if ((s & 1) && (t & 1)) val = -1.0e30f;
            sc[tn][v] = val;
        }
    }

    // ---- row softmax; shuffle reduce stays inside each 16-lane half ----
    #pragma unroll
    for (int v = 0; v < 8; ++v) {
        float m = -3.0e38f;
        #pragma unroll
        for (int tn = 0; tn < 8; ++tn) m = fmaxf(m, sc[tn][v]);
        #pragma unroll
        for (int off = 8; off >= 1; off >>= 1)
            m = fmaxf(m, __shfl_xor(m, off, 32));
        float ssum = 0.0f;
        #pragma unroll
        for (int tn = 0; tn < 8; ++tn) {
            const float p = __expf(sc[tn][v] - m);
            sc[tn][v] = p;
            ssum += p;
        }
        #pragma unroll
        for (int off = 8; off >= 1; off >>= 1)
            ssum += __shfl_xor(ssum, off, 32);
        const float inv = 1.0f / ssum;
        const int s = s0 + v + 8 * half;
        #pragma unroll
        for (int tn = 0; tn < 8; ++tn)
            sP[s][tn * 16 + l16] = (__bf16)(sc[tn][v] * inv);
    }
    __syncthreads();   // sVT + sP visible to everyone

    // ---- out = P V ----
    v8f oacc[4];
    #pragma unroll
    for (int t = 0; t < 4; ++t) oacc[t] = (v8f){};

    const __bf16* prow = sP[s0 + l16];
    #pragma unroll
    for (int t0 = 0; t0 < SEG_SZ; t0 += 32) {
        v16bf a = load_frag(prow + t0 + half * 16, prow + t0 + half * 16 + 8);
        #pragma unroll
        for (int dt = 0; dt < 4; ++dt) {
            const __bf16* vr = &sVT[dt * 16 + l16][t0 + half * 16];
            v16bf bb = load_frag(vr, vr + 8);
            oacc[dt] = wmma_bf16(a, bb, oacc[dt]);
        }
    }

    // ---- write y (bf16, stays in Hilbert order like the reference) ----
    #pragma unroll
    for (int dt = 0; dt < 4; ++dt) {
        const int d = dt * 16 + l16;
        #pragma unroll
        for (int v = 0; v < 8; ++v) {
            const int s = s0 + v + 8 * half;
            y[(row0 + s) * HID_SZ + h * HD_SZ + d] = (__bf16)oacc[dt][v];
        }
    }
}

// ---------------- host launcher ----------------
extern "C" void kernel_launch(void* const* d_in, const int* in_sizes, int n_in,
                              void* d_out, int out_size, void* d_ws, size_t ws_size,
                              hipStream_t stream) {
    const float* x     = (const float*)d_in[0];   // (4,4096,1024) f32
    const float* w_qkv = (const float*)d_in[1];   // (3072,1024)   f32
    const float* w_out = (const float*)d_in[2];   // (1024,1024)   f32
    const int*   hmap  = (const int*)  d_in[3];   // (4096)        i32

    char* ws = (char*)d_ws;
    size_t off = 0;
    __bf16* x_b    = (__bf16*)(ws + off); off += (size_t)ROWS * HID_SZ * 2;   // 32 MB
    __bf16* wqkv_b = (__bf16*)(ws + off); off += (size_t)QKV_N * HID_SZ * 2;  //  6 MB
    __bf16* wout_b = (__bf16*)(ws + off); off += (size_t)HID_SZ * HID_SZ * 2; //  2 MB
    __bf16* qkv_b  = (__bf16*)(ws + off); off += (size_t)ROWS * QKV_N * 2;    // 96 MB
    __bf16* y_b    = (__bf16*)(ws + off); off += (size_t)ROWS * HID_SZ * 2;   // 32 MB
    (void)ws_size; (void)in_sizes; (void)n_in; (void)out_size;

    cvt_f32_to_bf16<<<4096, 256, 0, stream>>>(x,     x_b,    ROWS * HID_SZ);
    cvt_f32_to_bf16<<<2048, 256, 0, stream>>>(w_qkv, wqkv_b, QKV_N * HID_SZ);
    cvt_f32_to_bf16<<<1024, 256, 0, stream>>>(w_out, wout_b, HID_SZ * HID_SZ);

    // qkv (Hilbert-permuted rows), bf16 out: (16384/128) * (3072/64) = 6144 blocks
    gemm_nt_bf16<<<(ROWS / 128) * (QKV_N / 64), 256, 0, stream>>>(
        x_b, wqkv_b, nullptr, qkv_b, hmap, ROWS, QKV_N, HID_SZ);

    // attention: one block per (b,h,seg)
    hilbert_attn<<<B_SZ * NH * NSEG, 256, 0, stream>>>(qkv_b, y_b);

    // final projection, fp32 out: (16384/128) * (1024/64) = 2048 blocks
    gemm_nt_bf16<<<(ROWS / 128) * (HID_SZ / 64), 256, 0, stream>>>(
        y_b, wout_b, (float*)d_out, nullptr, nullptr, ROWS, HID_SZ, HID_SZ);
}
